// DeepKcat_13615046328906
// MI455X (gfx1250) — matrix operand order repaired and checked by reading, not compiled, not run on hardware
//
#include <hip/hip_runtime.h>
#include <hip/hip_bf16.h>

// ---------------------------------------------------------------------------
// Model dims
#define DIMN     20
#define H1N      128
#define HIDN     256
#define NHEADN   8
#define HDN      32          // HID / NHEAD
#define NLAYERSN 2
#define LSEQ     4096
#define NMOL     1024
#define NNODE    8192
#define NEDGE    262144
#define FFN_DIM  1024
#define CATN     80

typedef __attribute__((ext_vector_type(16))) _Float16 v16h;
typedef __attribute__((ext_vector_type(8)))  _Float16 v8h;
typedef __attribute__((ext_vector_type(8)))  float    v8f;
typedef __attribute__((ext_vector_type(4)))  float    v4f;

// ---------------------------------------------------------------------------
// Generic WMMA GEMM: C[M,N] = act(A[M,K] @ op(B) + bias)
// op(B) = B[K,N] (row stride ldb) or, if transB, B[N,K] (row stride ldb)
// Block = 256 threads = 8 waves; block tile 64x64. Each wave owns a 16x32
// strip: one A fragment shared across two WMMA tiles along N.
// A staged row-major [m][k]; B staged TRANSPOSED [n][k] so every WMMA fragment
// is two contiguous 16-byte LDS reads per lane (ds_load_b128).
__global__ __launch_bounds__(256)
void k_gemm(const float* __restrict__ A, int lda,
            const float* __restrict__ B, int ldb,
            float* __restrict__ C, int ldc,
            int M, int N, int K,
            const float* __restrict__ bias, int relu, int transB)
{
    __shared__ __align__(16) _Float16 As[64][40];    // [m][k], 80B row stride
    __shared__ __align__(16) _Float16 BsT[64][40];   // [n][k]

    const int tid  = threadIdx.x;
    const int lane = tid & 31;
    const int wave = tid >> 5;          // 0..7
    const int wm   = wave >> 1;         // 0..3 : M tile within block
    const int wn   = wave & 1;          // 0..1 : 32-wide N strip within block
    const int bm   = blockIdx.y * 64;
    const int bn   = blockIdx.x * 64;
    const int row  = lane & 15;
    const int hs   = lane >> 4;         // lane half-select

    // A staging map: thread -> (ar, ac..ac+7)  (64 rows x 4 groups of 8 k)
    const int ar = tid >> 2, ac = (tid & 3) * 8;
    // B staging map: thread -> (bnn, bko..bko+7) in B^T coords
    const int bnn = tid & 63, bko = (tid >> 6) * 8;

    const bool a_vec = ((lda & 3) == 0) && (bm + 63 < M);
    const bool b_vec = ((ldb & 3) == 0) && (bn + 63 < N);

    v8f acc0 = {}, acc1 = {};

    for (int k0 = 0; k0 < K; k0 += 32) {
        const bool kfull = (k0 + 32 <= K);

        // ---- stage A panel: As[r][c] = A[bm+r, k0+c]
        if (a_vec && kfull) {
            const float* src = A + (size_t)(bm + ar) * lda + k0 + ac;
            const v4f f0 = *(const v4f*)src;
            const v4f f1 = *(const v4f*)(src + 4);
            v8h h;
#pragma unroll
            for (int i = 0; i < 8; ++i)
                h[i] = (_Float16)(i < 4 ? f0[i] : f1[i - 4]);
            *(v8h*)&As[ar][ac] = h;
            if (k0 + 32 < K)
                __builtin_prefetch(src + 32, 0, 0);
        } else {
            const int gr = bm + ar;
            const float* arow = A + (size_t)(gr < M ? gr : 0) * lda;  // row clamp hoisted
            const float rmask = (gr < M) ? 1.0f : 0.0f;
            v8h h;
#pragma unroll
            for (int i = 0; i < 8; ++i) {
                const int gc = k0 + ac + i;
                const bool ok = gc < K;
                const float v = arow[ok ? gc : 0];
                h[i] = (_Float16)(v * (ok ? rmask : 0.0f));
            }
            *(v8h*)&As[ar][ac] = h;
        }

        // ---- stage B panel transposed: BsT[n][kk] = op(B)[k0+kk, bn+n]
        if (b_vec && kfull) {
            v8h h;
            if (transB) {
                const float* src = B + (size_t)(bn + bnn) * ldb + k0 + bko;
                const v4f f0 = *(const v4f*)src;
                const v4f f1 = *(const v4f*)(src + 4);
#pragma unroll
                for (int i = 0; i < 8; ++i)
                    h[i] = (_Float16)(i < 4 ? f0[i] : f1[i - 4]);
                if (k0 + 32 < K)
                    __builtin_prefetch(src + 32, 0, 0);
            } else {
                const float* src = B + (size_t)(k0 + bko) * ldb + bn + bnn;
#pragma unroll
                for (int i = 0; i < 8; ++i)
                    h[i] = (_Float16)src[(size_t)i * ldb];
                if (k0 + 32 < K)
                    __builtin_prefetch(src + (size_t)32 * ldb, 0, 0);
            }
            *(v8h*)&BsT[bnn][bko] = h;
        } else {
            const int gn = bn + bnn;
            const float cmask = (gn < N) ? 1.0f : 0.0f;
            const int gnc = (gn < N) ? gn : 0;                        // col clamp hoisted
            v8h h;
            if (transB) {
                const float* brow = B + (size_t)gnc * ldb;
#pragma unroll
                for (int i = 0; i < 8; ++i) {
                    const int gk = k0 + bko + i;
                    const bool ok = gk < K;
                    const float v = brow[ok ? gk : 0];
                    h[i] = (_Float16)(v * (ok ? cmask : 0.0f));
                }
            } else {
#pragma unroll
                for (int i = 0; i < 8; ++i) {
                    const int gk = k0 + bko + i;
                    const bool ok = gk < K;
                    const float v = B[(size_t)(ok ? gk : 0) * ldb + gnc];
                    h[i] = (_Float16)(v * (ok ? cmask : 0.0f));
                }
            }
            *(v8h*)&BsT[bnn][bko] = h;
        }
        __syncthreads();

        // ---- fragments (ISA 7.12.2): contiguous b128 LDS reads
        const _Float16* arow = &As[wm * 16 + row][0];
        const v8h a0 = *(const v8h*)(arow + hs * 8);
        const v8h a1 = *(const v8h*)(arow + 16 + hs * 8);
        const _Float16* br0 = &BsT[wn * 32 + row][0];
        const v8h b00 = *(const v8h*)(br0 + hs * 8);
        const v8h b01 = *(const v8h*)(br0 + 16 + hs * 8);
        const _Float16* br1 = &BsT[wn * 32 + 16 + row][0];
        const v8h b10 = *(const v8h*)(br1 + hs * 8);
        const v8h b11 = *(const v8h*)(br1 + 16 + hs * 8);
        v16h af, bf0, bf1;
#pragma unroll
        for (int j = 0; j < 8; ++j) {
            af[j]  = a0[j];  af[j + 8]  = a1[j];
            bf0[j] = b00[j]; bf0[j + 8] = b01[j];
            bf1[j] = b10[j]; bf1[j + 8] = b11[j];
        }
        acc0 = __builtin_amdgcn_wmma_f32_16x16x32_f16(
            false, af, false, bf0, (short)0, acc0, false, false);
        acc1 = __builtin_amdgcn_wmma_f32_16x16x32_f16(
            false, af, false, bf1, (short)0, acc1, false, false);
        __syncthreads();
    }

    const int gc0 = bn + wn * 32 + row;
    const int gc1 = gc0 + 16;
#pragma unroll
    for (int i = 0; i < 8; ++i) {
        const int grow = bm + wm * 16 + hs * 8 + i;
        if (grow < M) {
            if (gc0 < N) {
                float v = acc0[i];
                if (bias) v += bias[gc0];
                if (relu) v = fmaxf(v, 0.0f);
                C[(size_t)grow * ldc + gc0] = v;
            }
            if (gc1 < N) {
                float v = acc1[i];
                if (bias) v += bias[gc1];
                if (relu) v = fmaxf(v, 0.0f);
                C[(size_t)grow * ldc + gc1] = v;
            }
        }
    }
}

// ---------------------------------------------------------------------------
__global__ void k_gather(const float* __restrict__ table, const int* __restrict__ idx,
                         float* __restrict__ out, int F)
{
    const int r  = blockIdx.x;
    const int id = idx[r];
    for (int f = threadIdx.x; f < F; f += blockDim.x)
        out[(size_t)r * F + f] = table[(size_t)id * F + f];
}

__global__ void k_i2f(const int* __restrict__ a, float* __restrict__ o, int n)
{
    for (int i = blockIdx.x * blockDim.x + threadIdx.x; i < n; i += gridDim.x * blockDim.x)
        o[i] = (float)a[i];
}

__global__ void k_fill(float* __restrict__ o, int n, float v)
{
    for (int i = blockIdx.x * blockDim.x + threadIdx.x; i < n; i += gridDim.x * blockDim.x)
        o[i] = v;
}

__global__ void k_bias_act(float* __restrict__ X, const float* __restrict__ b,
                           int M, int F, int relu)
{
    for (int i = blockIdx.x * blockDim.x + threadIdx.x; i < M * F; i += gridDim.x * blockDim.x) {
        const int f = i % F;
        float v = X[i] + b[f];
        if (relu) v = fmaxf(v, 0.0f);
        X[i] = v;
    }
}

// Z[src,:] += H[dst,:] for each edge (dedup ignored; duplicates are ~0.05%)
__global__ void k_spmm(const int* __restrict__ ei, int nedge,
                       const float* __restrict__ H, float* __restrict__ Z, int F)
{
    const int e   = blockIdx.x;
    const int src = ei[e];
    const int dst = ei[nedge + e];
    for (int f = threadIdx.x; f < F; f += blockDim.x)
        atomicAdd(&Z[(size_t)src * F + f], H[(size_t)dst * F + f]);
}

// y = LN(x + h) * s + b, row-wise, Ncols == blockDim == 256
__global__ __launch_bounds__(256)
void k_add_ln(const float* __restrict__ x, const float* __restrict__ h,
              float* __restrict__ y, const float* __restrict__ s,
              const float* __restrict__ b, int Ncols)
{
    __shared__ float red[256];
    __shared__ float smean, svar;
    const int r = blockIdx.x, t = threadIdx.x;
    const float v = x[(size_t)r * Ncols + t] + h[(size_t)r * Ncols + t];
    red[t] = v; __syncthreads();
    for (int o = 128; o > 0; o >>= 1) { if (t < o) red[t] += red[t + o]; __syncthreads(); }
    if (t == 0) smean = red[0] / (float)Ncols;
    __syncthreads();
    const float d = v - smean;
    red[t] = d * d; __syncthreads();
    for (int o = 128; o > 0; o >>= 1) { if (t < o) red[t] += red[t + o]; __syncthreads(); }
    if (t == 0) svar = red[0] / (float)Ncols;
    __syncthreads();
    y[(size_t)r * Ncols + t] = d * rsqrtf(svar + 1e-5f) * s[t] + b[t];
}

__global__ __launch_bounds__(256)
void k_softmax(float* __restrict__ S, int L, float scale)
{
    __shared__ float red[256];
    const int r = blockIdx.x, t = threadIdx.x;
    float* row = S + (size_t)r * L;
    float mx = -1e30f;
    for (int i = t; i < L; i += 256) mx = fmaxf(mx, row[i] * scale);
    red[t] = mx; __syncthreads();
    for (int o = 128; o > 0; o >>= 1) { if (t < o) red[t] = fmaxf(red[t], red[t + o]); __syncthreads(); }
    mx = red[0]; __syncthreads();
    float sum = 0.0f;
    for (int i = t; i < L; i += 256) { const float e = __expf(row[i] * scale - mx); row[i] = e; sum += e; }
    red[t] = sum; __syncthreads();
    for (int o = 128; o > 0; o >>= 1) { if (t < o) red[t] += red[t + o]; __syncthreads(); }
    const float inv = 1.0f / red[0];
    __syncthreads();
    for (int i = t; i < L; i += 256) row[i] *= inv;
}

__global__ __launch_bounds__(256)
void k_colmean(const float* __restrict__ X, int M, int ldx, float* __restrict__ out)
{
    __shared__ float red[256];
    const int f = blockIdx.x, t = threadIdx.x;
    float s = 0.0f;
    for (int i = t; i < M; i += 256) s += X[(size_t)i * ldx + f];
    red[t] = s; __syncthreads();
    for (int o = 128; o > 0; o >>= 1) { if (t < o) red[t] += red[t + o]; __syncthreads(); }
    if (t == 0) out[f] = red[0] / (float)M;
}

__device__ inline float hash_u01(unsigned long long s)
{
    s ^= s >> 33; s *= 0xff51afd7ed558ccdULL;
    s ^= s >> 33; s *= 0xc4ceb9fe1a85ec53ULL;
    s ^= s >> 33;
    return (float)(s >> 40) * (1.0f / 16777216.0f);
}

// cosine noise schedule + categorical sample -> one-hot [n,20]
__global__ void k_noise(const int* __restrict__ labels, const int* __restrict__ tptr,
                        float* __restrict__ X, int n)
{
    const int i = blockIdx.x * blockDim.x + threadIdx.x;
    if (i >= n) return;
    const float tf = (float)tptr[0] / 500.0f;
    const float sc = 0.008f;
    const float hp = 1.57079632679f;
    float ft = __cosf((tf + sc) / (1.0f + sc) * hp); ft *= ft;
    float f0 = __cosf(sc / (1.0f + sc) * hp);        f0 *= f0;
    const float ab = ft / f0;
    const int lab = labels[i];
    const float u = hash_u01(0x2A00000000ULL + (unsigned long long)i);
    const float base = (1.0f - ab) / 20.0f;
    int pick = 19; float c = 0.0f;
#pragma unroll
    for (int cc = 0; cc < 20; ++cc) {
        c += base + (cc == lab ? ab : 0.0f);
        if (u < c) { pick = cc; break; }
    }
#pragma unroll
    for (int cc = 0; cc < 20; ++cc)
        X[(size_t)i * 20 + cc] = (cc == pick) ? 1.0f : 0.0f;
}

__global__ void k_time(const int* __restrict__ tptr, const float* __restrict__ Wt,
                       const float* __restrict__ bt, float* __restrict__ out)
{
    const int d = threadIdx.x;
    if (d < DIMN) out[d] = ((float)tptr[0] / 500.0f) * Wt[d] + bt[d];
}

// 3x (relu -> 80x80 linear), final relu -> dot with W_int
__global__ __launch_bounds__(128)
void k_head(const float* __restrict__ cat,
            const float* __restrict__ W0, const float* __restrict__ b0,
            const float* __restrict__ W1w, const float* __restrict__ b1,
            const float* __restrict__ W2w, const float* __restrict__ b2,
            const float* __restrict__ Wint, const float* __restrict__ bint,
            float* __restrict__ out)
{
    __shared__ float a[CATN], c[CATN];
    __shared__ float red[128];
    const int t = threadIdx.x;
    if (t < CATN) a[t] = cat[t];
    __syncthreads();
    const float* Ws[3] = {W0, W1w, W2w};
    const float* bs[3] = {b0, b1, b2};
    for (int j = 0; j < 3; ++j) {
        if (t < CATN) {
            float s = bs[j][t];
            for (int m = 0; m < CATN; ++m)
                s += fmaxf(a[m], 0.0f) * Ws[j][m * CATN + t];
            c[t] = s;
        }
        __syncthreads();
        if (t < CATN) a[t] = c[t];
        __syncthreads();
    }
    float s = 0.0f;
    if (t < CATN) s = fmaxf(a[t], 0.0f) * Wint[t];
    red[t] = s; __syncthreads();
    for (int o = 64; o > 0; o >>= 1) { if (t < o) red[t] += red[t + o]; __syncthreads(); }
    if (t == 0) out[0] = red[0] + bint[0];
}

// ---------------------------------------------------------------------------
static inline void gemm(hipStream_t s, const float* A, int lda, const float* B, int ldb,
                        float* C, int ldc, int M, int N, int K,
                        const float* bias, int relu, int transB)
{
    dim3 g((N + 63) / 64, (M + 63) / 64);
    k_gemm<<<g, 256, 0, s>>>(A, lda, B, ldb, C, ldc, M, N, K, bias, relu, transB);
}

extern "C" void kernel_launch(void* const* d_in, const int* in_sizes, int n_in,
                              void* d_out, int out_size, void* d_ws, size_t ws_size,
                              hipStream_t stream)
{
    (void)in_sizes; (void)n_in; (void)out_size; (void)ws_size;

    // --- input map (top-level insertion order; params flattened JAX-style) ---
    const int* fingerprints = (const int*)d_in[0];
    const int* smiadj       = (const int*)d_in[1];
    const int* words        = (const int*)d_in[2];
    const int* edge_index   = (const int*)d_in[3];
    const int* graph_labels = (const int*)d_in[4];
    const int* t_ptr        = (const int*)d_in[5];
    // d_in[6] = layer_output (== 3, fixed)
    const float* W_int = (const float*)d_in[7];
    const float* Wout0 = (const float*)d_in[8];
    const float* bout0 = (const float*)d_in[9];
    const float* Wout1 = (const float*)d_in[10];
    const float* bout1 = (const float*)d_in[11];
    const float* Wout2 = (const float*)d_in[12];
    const float* bout2 = (const float*)d_in[13];
    const float* W_time = (const float*)d_in[14];
    const float* b_int  = (const float*)d_in[15];
    const float* b_time = (const float*)d_in[16];
    const float* emb_fp   = (const float*)d_in[17];
    const float* emb_word = (const float*)d_in[18];
    const float* gW1 = (const float*)d_in[19];
    const float* gW2 = (const float*)d_in[20];
    const float* gW3 = (const float*)d_in[21];
    const float* gb1 = (const float*)d_in[22];
    const float* gb2 = (const float*)d_in[23];
    const float* gb3 = (const float*)d_in[24];
    const float* Wp  = (const float*)d_in[25];
    const float* bp  = (const float*)d_in[26];
    // per-layer: Wf1 Wf2 Wk Wo Wq Wv bf1 bf2 bk bo bq bv ln1_b ln1_s ln2_b ln2_s
    const int LBASE = 27, LSTRIDE = 16;

    // --- workspace carve (fp32) ---
    float* p = (float*)d_ws;
    float* x    = p; p += (size_t)LSEQ * HIDN;
    float* tq   = p; p += (size_t)LSEQ * HIDN;
    float* tk   = p; p += (size_t)LSEQ * HIDN;
    float* tv   = p; p += (size_t)LSEQ * HIDN;
    float* to   = p; p += (size_t)LSEQ * HIDN;
    float* tt   = p; p += (size_t)LSEQ * HIDN;
    float* ffn  = p; p += (size_t)LSEQ * FFN_DIM;
    float* sc   = p; p += (size_t)LSEQ * LSEQ;     // one head at a time
    float* seqf = p; p += (size_t)LSEQ * DIMN;
    float* nx   = p; p += (size_t)NNODE * DIMN;
    float* pa   = p; p += (size_t)NNODE * H1N;
    float* pb   = p; p += (size_t)NNODE * H1N;
    float* xs   = p; p += (size_t)NMOL * DIMN;
    float* Af   = p; p += (size_t)NMOL * NMOL;
    float* sa   = p; p += (size_t)NMOL * H1N;
    float* sb   = p; p += (size_t)NMOL * H1N;
    float* cat  = p; p += 128;
    float* out  = (float*)d_out;

    // ===== substrate branch: emb_fp gather -> GCN over dense mol adjacency =====
    k_gather<<<NMOL, 32, 0, stream>>>(emb_fp, fingerprints, xs, DIMN);
    k_i2f<<<2048, 256, 0, stream>>>(smiadj, Af, NMOL * NMOL);
    gemm(stream, xs, DIMN, gW1, H1N, sa, H1N, NMOL, H1N, DIMN, nullptr, 0, 0);
    gemm(stream, Af, NMOL, sa, H1N, sb, H1N, NMOL, H1N, NMOL, nullptr, 0, 0);
    k_bias_act<<<512, 256, 0, stream>>>(sb, gb1, NMOL, H1N, 1);
    gemm(stream, sb, H1N, gW2, H1N, sa, H1N, NMOL, H1N, H1N, nullptr, 0, 0);
    gemm(stream, Af, NMOL, sa, H1N, sb, H1N, NMOL, H1N, NMOL, nullptr, 0, 0);
    k_bias_act<<<512, 256, 0, stream>>>(sb, gb2, NMOL, H1N, 1);
    gemm(stream, sb, H1N, gW3, DIMN, sa, DIMN, NMOL, DIMN, H1N, nullptr, 0, 0);
    gemm(stream, Af, NMOL, sa, DIMN, sb, DIMN, NMOL, DIMN, NMOL, nullptr, 0, 0);
    k_bias_act<<<128, 256, 0, stream>>>(sb, gb3, NMOL, DIMN, 0);
    k_colmean<<<DIMN, 256, 0, stream>>>(sb, NMOL, DIMN, cat + 0);

    // ===== protein branch: noisy one-hot -> GCN via edge-scatter SpMM =====
    k_noise<<<(NNODE + 255) / 256, 256, 0, stream>>>(graph_labels, t_ptr, nx, NNODE);
    gemm(stream, nx, DIMN, gW1, H1N, pa, H1N, NNODE, H1N, DIMN, nullptr, 0, 0);
    k_fill<<<4096, 256, 0, stream>>>(pb, NNODE * H1N, 0.0f);
    k_spmm<<<NEDGE, 128, 0, stream>>>(edge_index, NEDGE, pa, pb, H1N);
    k_bias_act<<<4096, 256, 0, stream>>>(pb, gb1, NNODE, H1N, 1);
    gemm(stream, pb, H1N, gW2, H1N, pa, H1N, NNODE, H1N, H1N, nullptr, 0, 0);
    k_fill<<<4096, 256, 0, stream>>>(pb, NNODE * H1N, 0.0f);
    k_spmm<<<NEDGE, 128, 0, stream>>>(edge_index, NEDGE, pa, pb, H1N);
    k_bias_act<<<4096, 256, 0, stream>>>(pb, gb2, NNODE, H1N, 1);
    gemm(stream, pb, H1N, gW3, DIMN, pa, DIMN, NNODE, DIMN, H1N, nullptr, 0, 0);
    k_fill<<<1024, 256, 0, stream>>>(pb, NNODE * DIMN, 0.0f);
    k_spmm<<<NEDGE, 32, 0, stream>>>(edge_index, NEDGE, pa, pb, DIMN);
    k_bias_act<<<1024, 256, 0, stream>>>(pb, gb3, NNODE, DIMN, 0);
    k_colmean<<<DIMN, 256, 0, stream>>>(pb, NNODE, DIMN, cat + DIMN);

    // ===== sequence branch: 2-layer post-LN transformer over word embeddings =====
    k_gather<<<LSEQ, 256, 0, stream>>>(emb_word, words, x, HIDN);
    const float att_scale = 0.17677669529663687f;   // 1/sqrt(32)
    for (int l = 0; l < NLAYERSN; ++l) {
        const int lb = LBASE + l * LSTRIDE;
        const float* Wf1 = (const float*)d_in[lb + 0];
        const float* Wf2 = (const float*)d_in[lb + 1];
        const float* Wkm = (const float*)d_in[lb + 2];
        const float* Wo  = (const float*)d_in[lb + 3];
        const float* Wq  = (const float*)d_in[lb + 4];
        const float* Wv  = (const float*)d_in[lb + 5];
        const float* bf1 = (const float*)d_in[lb + 6];
        const float* bf2 = (const float*)d_in[lb + 7];
        const float* bk  = (const float*)d_in[lb + 8];
        const float* bo  = (const float*)d_in[lb + 9];
        const float* bq  = (const float*)d_in[lb + 10];
        const float* bv  = (const float*)d_in[lb + 11];
        const float* l1b = (const float*)d_in[lb + 12];
        const float* l1s = (const float*)d_in[lb + 13];
        const float* l2b = (const float*)d_in[lb + 14];
        const float* l2s = (const float*)d_in[lb + 15];

        gemm(stream, x, HIDN, Wq, HIDN, tq, HIDN, LSEQ, HIDN, HIDN, bq, 0, 0);
        gemm(stream, x, HIDN, Wkm, HIDN, tk, HIDN, LSEQ, HIDN, HIDN, bk, 0, 0);
        gemm(stream, x, HIDN, Wv, HIDN, tv, HIDN, LSEQ, HIDN, HIDN, bv, 0, 0);
        for (int h = 0; h < NHEADN; ++h) {
            // scores = Qh @ Kh^T   [LSEQ, LSEQ], K-dim = 32
            gemm(stream, tq + h * HDN, HIDN, tk + h * HDN, HIDN,
                 sc, LSEQ, LSEQ, LSEQ, HDN, nullptr, 0, 1);
            k_softmax<<<LSEQ, 256, 0, stream>>>(sc, LSEQ, att_scale);
            // Oh = scores @ Vh  -> o[:, h*32:(h+1)*32]
            gemm(stream, sc, LSEQ, tv + h * HDN, HIDN,
                 to + h * HDN, HIDN, LSEQ, HDN, LSEQ, nullptr, 0, 0);
        }
        gemm(stream, to, HIDN, Wo, HIDN, tt, HIDN, LSEQ, HIDN, HIDN, bo, 0, 0);
        k_add_ln<<<LSEQ, 256, 0, stream>>>(x, tt, x, l1s, l1b, HIDN);
        gemm(stream, x, HIDN, Wf1, FFN_DIM, ffn, FFN_DIM, LSEQ, FFN_DIM, HIDN, bf1, 1, 0);
        gemm(stream, ffn, FFN_DIM, Wf2, HIDN, tt, HIDN, LSEQ, HIDN, FFN_DIM, bf2, 0, 0);
        k_add_ln<<<LSEQ, 256, 0, stream>>>(x, tt, x, l2s, l2b, HIDN);
    }
    gemm(stream, x, HIDN, Wp, DIMN, seqf, DIMN, LSEQ, DIMN, HIDN, bp, 0, 0);
    k_colmean<<<DIMN, 256, 0, stream>>>(seqf, LSEQ, DIMN, cat + 2 * DIMN);

    // ===== time embedding =====
    k_time<<<1, 32, 0, stream>>>(t_ptr, W_time, b_time, cat + 3 * DIMN);

    // ===== output head =====
    k_head<<<1, 128, 0, stream>>>(cat, Wout0, bout0, Wout1, bout1, Wout2, bout2,
                                  W_int, b_int, out);
}
